// EncoderBlock_28432683499827
// MI455X (gfx1250) — matrix operand order repaired and checked by reading, not compiled, not run on hardware
//
#include <hip/hip_runtime.h>
#include <math.h>

// Problem constants (fixed by the reference)
#define B_   2
#define S_   2048
#define D_   1024
#define H_   16
#define DK_  64
#define DFF_ 4096
#define MTOK (B_ * S_)   // 4096 token rows

typedef __attribute__((ext_vector_type(16))) _Float16 v16h;
typedef __attribute__((ext_vector_type(8)))  _Float16 v8h;
typedef __attribute__((ext_vector_type(8)))  float    v8f;

// ---------------------------------------------------------------------------
// WMMA helpers (CDNA5 wave32, 16x16x32 f16 -> f32 accumulate)
// ---------------------------------------------------------------------------
__device__ __forceinline__ v8f wmma_f16(v16h a, v16h b, v8f c) {
  // 8 args: (neg_a, A, neg_b, B, c_mod, C, reuse_a, reuse_b)
  return __builtin_amdgcn_wmma_f32_16x16x32_f16(false, a, false, b, (short)0, c,
                                                false, false);
}

// A-fragment (16x32, 16-bit): per lane 16 contiguous K values starting at
// k=(lane/16)*16 (caller passes base+(lane/16)*16 address).
__device__ __forceinline__ v16h load_frag_a(const _Float16* p) {
  union { v16h v; v8h h[2]; } u;
  u.h[0] = *(const v8h*)(p);
  u.h[1] = *(const v8h*)(p + 8);
  return u.v;
}

// B-fragment (32x16, 16-bit): per lane 16 contiguous K values starting at
// (lane/16)*16 (operands stored so K is contiguous).
__device__ __forceinline__ v16h load_frag_b(const _Float16* p) {
  union { v16h v; v8h h[2]; } u;
  u.h[0] = *(const v8h*)(p);
  u.h[1] = *(const v8h*)(p + 8);
  return u.v;
}

// ---------------------------------------------------------------------------
// CDNA5 async global->LDS copies (ASYNCcnt), see CDNA5 ISA.
// Low 32 bits of a generic pointer to __shared__ == LDS byte offset (flat
// aperture rule, ISA 10.2), which is what VDST of the async load wants.
// ---------------------------------------------------------------------------
__device__ __forceinline__ void async_copy_b128(void* lds_dst, const void* gsrc) {
  uint32_t loff = (uint32_t)(uintptr_t)lds_dst;
  asm volatile("global_load_async_to_lds_b128 %0, %1, off"
               :
               : "v"(loff), "v"(gsrc)
               : "memory");
}
__device__ __forceinline__ void wait_async_le6() {
  asm volatile("s_wait_asynccnt 6" ::: "memory");
}
__device__ __forceinline__ void wait_async_le4() {
  asm volatile("s_wait_asynccnt 4" ::: "memory");
}
__device__ __forceinline__ void wait_async_0() {
  asm volatile("s_wait_asynccnt 0" ::: "memory");
}

// ---------------------------------------------------------------------------
// Weight conversion: fp32 -> f16, transposed to [N, K] so B-fragments are
// contiguous-in-K per lane.
// ---------------------------------------------------------------------------
__global__ void w_transpose_f16(const float* __restrict__ W,
                                _Float16* __restrict__ Wt, int K, int N) {
  size_t idx = (size_t)blockIdx.x * 256 + threadIdx.x;  // idx = n*K + k
  if (idx >= (size_t)N * K) return;
  int n = (int)(idx / K);
  int k = (int)(idx % K);
  Wt[idx] = (_Float16)W[(size_t)k * N + n];
}

// Stacked per-head QKV weight [H, D, DK] -> [H*DK, D] f16 (n = h*DK+dk, k = d)
__global__ void qkv_w_transpose_f16(const float* __restrict__ W,
                                    _Float16* __restrict__ Wt) {
  size_t idx = (size_t)blockIdx.x * 256 + threadIdx.x;  // idx = n*D + d
  if (idx >= (size_t)D_ * D_) return;
  int n = (int)(idx >> 10);        // / D_
  int d = (int)(idx & (D_ - 1));
  int h  = n >> 6;                 // / DK_
  int dk = n & (DK_ - 1);
  Wt[idx] = (_Float16)W[((size_t)h * D_ + d) * DK_ + dk];
}

// ---------------------------------------------------------------------------
// LayerNorm: fp32 row -> f16 row (GEMM A operand)
// ---------------------------------------------------------------------------
__global__ __launch_bounds__(256)
void ln_f16_kernel(const float* __restrict__ x, const float* __restrict__ g,
                   const float* __restrict__ b, _Float16* __restrict__ out) {
  __shared__ float s1[256], s2[256];
  const int row = blockIdx.x;
  const float* xr = x + (size_t)row * D_;
  float lsum = 0.f, lsq = 0.f;
  for (int i = threadIdx.x; i < D_; i += 256) {
    float v = xr[i];
    lsum += v;
    lsq  += v * v;
  }
  s1[threadIdx.x] = lsum;
  s2[threadIdx.x] = lsq;
  __syncthreads();
  for (int off = 128; off > 0; off >>= 1) {
    if ((int)threadIdx.x < off) {
      s1[threadIdx.x] += s1[threadIdx.x + off];
      s2[threadIdx.x] += s2[threadIdx.x + off];
    }
    __syncthreads();
  }
  float mu  = s1[0] * (1.f / D_);
  float var = s2[0] * (1.f / D_) - mu * mu;
  float rs  = rsqrtf(var + 1e-5f);
  _Float16* orow = out + (size_t)row * D_;
  for (int i = threadIdx.x; i < D_; i += 256)
    orow[i] = (_Float16)((xr[i] - mu) * rs * g[i] + b[i]);
}

// ---------------------------------------------------------------------------
// WMMA GEMM with double-buffered async global->LDS staging.
// C[M,N] = A[M,K] * Bt[N,K]^T + bias, fused epilogues.
// Block = 256 threads = 8 waves; block tile 128M x 64N x 64K (two 32-K
// sub-steps per stage => 8 WMMAs/wave per barrier interval).
// LDS: A 128x64 + B 64x64, double buffered = 48 KB. Grid: (N/64, M/128).
// ---------------------------------------------------------------------------
enum { EPI_QK = 0, EPI_VT = 1, EPI_RES = 2, EPI_RELU = 3 };

template <int EPI>
__global__ __launch_bounds__(256)
void gemm_f16_kernel(const _Float16* __restrict__ A,
                     const _Float16* __restrict__ Bt,
                     const float* __restrict__ bias,
                     const float* __restrict__ resid,
                     void* __restrict__ out, int M, int N, int K) {
  __shared__ __attribute__((aligned(16))) _Float16 Asm[2][128][64];
  __shared__ __attribute__((aligned(16))) _Float16 Bsm[2][64][64];

  const int tid  = threadIdx.x;
  const int lane = tid & 31;
  const int wave = tid >> 5;
  const int hl = lane >> 4;     // half-wave select
  const int ln = lane & 15;
  const int mblk = (int)blockIdx.y * 128;
  const int n0   = (int)blockIdx.x * 64;

  // Staging: tiles split into 16-byte chunks (8 f16), 8 chunks per 64-K row.
  // A tile: 1024 chunks -> thread copies rows ar, ar+32, ar+64, ar+96.
  // B tile:  512 chunks -> thread copies rows ar, ar+32.
  const int ar   = tid >> 3;          // 0..31
  const int akof = (tid & 7) * 8;     // 0..56

  const _Float16* Ag = A + (size_t)(mblk + ar) * K + akof;
  const _Float16* Bg = Bt + (size_t)(n0 + ar) * K + akof;
  const size_t K32 = (size_t)K * 32;

  const int KT = K >> 6;  // k-stages of 64

  // Prologue: stage k-stage 0 into buffer 0 (6 async instructions per wave)
  async_copy_b128(&Asm[0][ar][akof],      Ag);
  async_copy_b128(&Asm[0][ar + 32][akof], Ag + K32);
  async_copy_b128(&Asm[0][ar + 64][akof], Ag + 2 * K32);
  async_copy_b128(&Asm[0][ar + 96][akof], Ag + 3 * K32);
  async_copy_b128(&Bsm[0][ar][akof],      Bg);
  async_copy_b128(&Bsm[0][ar + 32][akof], Bg + K32);

  v8f acc[4] = {};

  for (int kt = 0; kt < KT; ++kt) {
    const int cur = kt & 1;
    if (kt + 1 < KT) {
      const int nk = (kt + 1) << 6;
      async_copy_b128(&Asm[cur ^ 1][ar][akof],      Ag + nk);
      async_copy_b128(&Asm[cur ^ 1][ar + 32][akof], Ag + K32 + nk);
      async_copy_b128(&Asm[cur ^ 1][ar + 64][akof], Ag + 2 * K32 + nk);
      async_copy_b128(&Asm[cur ^ 1][ar + 96][akof], Ag + 3 * K32 + nk);
      async_copy_b128(&Bsm[cur ^ 1][ar][akof],      Bg + nk);
      async_copy_b128(&Bsm[cur ^ 1][ar + 32][akof], Bg + K32 + nk);
      wait_async_le6();   // 6 oldest (current stage) complete, next in flight
    } else {
      wait_async_0();
    }
    __syncthreads();      // current stage visible to all waves

#pragma unroll
    for (int kk = 0; kk < 2; ++kk) {
      v16h af = load_frag_a(&Asm[cur][wave * 16 + ln][kk * 32 + hl * 16]);
#pragma unroll
      for (int t = 0; t < 4; ++t) {
        v16h bf = load_frag_b(&Bsm[cur][t * 16 + ln][kk * 32 + hl * 16]);
        acc[t] = wmma_f16(af, bf, acc[t]);
      }
    }
    __syncthreads();      // all waves done reading before buffer is reloaded
  }

  // C/D layout: lane ln in half hl holds (row m0 + r + 8*hl, col n0+t*16+ln)
  const int m0 = mblk + wave * 16;
#pragma unroll
  for (int t = 0; t < 4; ++t) {
#pragma unroll
    for (int r = 0; r < 8; ++r) {
      const int m = m0 + r + 8 * hl;
      const int n = n0 + t * 16 + ln;
      float v = acc[t][r] + bias[n];
      if constexpr (EPI == EPI_QK) {
        int b = m >> 11, s = m & (S_ - 1);    // m = b*S + s
        int h = n >> 6,  dk = n & (DK_ - 1);  // n = h*DK + dk
        ((_Float16*)out)[(((size_t)b * H_ + h) * S_ + s) * DK_ + dk] = (_Float16)v;
      } else if constexpr (EPI == EPI_VT) {
        int b = m >> 11, s = m & (S_ - 1);
        int h = n >> 6,  dk = n & (DK_ - 1);
        ((_Float16*)out)[(((size_t)b * H_ + h) * DK_ + dk) * S_ + s] = (_Float16)v;
      } else if constexpr (EPI == EPI_RES) {
        ((float*)out)[(size_t)m * N + n] = v + resid[(size_t)m * N + n];
      } else {  // EPI_RELU
        ((_Float16*)out)[(size_t)m * N + n] = (_Float16)fmaxf(v, 0.f);
      }
    }
  }
}

// ---------------------------------------------------------------------------
// Flash attention with async-staged K/V panels shared by the block.
// q  : [B,H,S,DK] f16 (row-major per query -> A fragments, kept in VGPRs)
// k  : [B,H,S,DK] f16 -> staged to Ksm[32][64] per 32-key step
// vT : [B,H,DK,S] f16 -> staged to Vsm[64][32] per 32-key step
// ctx: [B,S,D]    f16 (heads concatenated, feeds Wo GEMM as A operand)
// Grid: (S/64, B*H), block = 128 (4 waves, all same (b,h)).
// ---------------------------------------------------------------------------
__global__ __launch_bounds__(128)
void flash_attn_kernel(const _Float16* __restrict__ q,
                       const _Float16* __restrict__ k,
                       const _Float16* __restrict__ vT,
                       _Float16* __restrict__ ctx) {
  __shared__ __attribute__((aligned(16))) _Float16 Ksm[2][32][64];  // 8 KB
  __shared__ __attribute__((aligned(16))) _Float16 Vsm[2][64][32];  // 8 KB
  __shared__ __attribute__((aligned(16))) _Float16 plds[4][16][32]; // 4 KB

  const int tid = threadIdx.x;
  const int lane = tid & 31, wave = tid >> 5;
  const int hl = lane >> 4, ln = lane & 15;
  const int bh = blockIdx.y;
  const int b = bh >> 4, h = bh & (H_ - 1);
  const int q0 = ((int)blockIdx.x * 4 + wave) * 16;

  const _Float16* qp = q  + ((size_t)bh * S_ + q0) * DK_;
  const _Float16* kp = k  + (size_t)bh * S_ * DK_;
  const _Float16* vp = vT + (size_t)bh * DK_ * S_;

  // Staging assignment (128 threads):
  // K panel 32x64 = 256 chunks -> rows kr, kr+16 ; V panel 64x32 = 256 chunks
  const int kr   = tid >> 3;          // 0..15
  const int kkof = (tid & 7) * 8;     // 0..56
  const int vr   = tid >> 2;          // 0..31
  const int vkof = (tid & 3) * 8;     // 0..24

  // Q fragments for dk 0..31 and 32..63 (held in VGPRs for the whole pass)
  v16h qf0 = load_frag_a(qp + (size_t)ln * DK_ + hl * 16);
  v16h qf1 = load_frag_a(qp + (size_t)ln * DK_ + 32 + hl * 16);

  // Prologue: stage keys/values 0..31 into buffer 0 (4 async per wave)
  async_copy_b128(&Ksm[0][kr][kkof],      kp + (size_t)kr * DK_ + kkof);
  async_copy_b128(&Ksm[0][kr + 16][kkof], kp + (size_t)(kr + 16) * DK_ + kkof);
  async_copy_b128(&Vsm[0][vr][vkof],      vp + (size_t)vr * S_ + vkof);
  async_copy_b128(&Vsm[0][vr + 32][vkof], vp + (size_t)(vr + 32) * S_ + vkof);

  v8f acc[4] = {};
  float ms[8], ls[8];
#pragma unroll
  for (int r = 0; r < 8; ++r) { ms[r] = -1e30f; ls[r] = 0.f; }

  for (int j = 0; j < S_; j += 32) {
    const int cur = (j >> 5) & 1;
    if (j + 32 < S_) {
      const int jn = j + 32;
      async_copy_b128(&Ksm[cur ^ 1][kr][kkof],
                      kp + (size_t)(jn + kr) * DK_ + kkof);
      async_copy_b128(&Ksm[cur ^ 1][kr + 16][kkof],
                      kp + (size_t)(jn + kr + 16) * DK_ + kkof);
      async_copy_b128(&Vsm[cur ^ 1][vr][vkof],
                      vp + (size_t)vr * S_ + jn + vkof);
      async_copy_b128(&Vsm[cur ^ 1][vr + 32][vkof],
                      vp + (size_t)(vr + 32) * S_ + jn + vkof);
      wait_async_le4();
    } else {
      wait_async_0();
    }
    __syncthreads();

    // --- scores for 32 keys: two 16x16 tiles, K-dim = DK in two WMMA steps
    v8f st0 = {}, st1 = {};
    st0 = wmma_f16(qf0, load_frag_b(&Ksm[cur][ln][hl * 16]), st0);
    st0 = wmma_f16(qf1, load_frag_b(&Ksm[cur][ln][32 + hl * 16]), st0);
    st1 = wmma_f16(qf0, load_frag_b(&Ksm[cur][16 + ln][hl * 16]), st1);
    st1 = wmma_f16(qf1, load_frag_b(&Ksm[cur][16 + ln][32 + hl * 16]), st1);

    // --- online softmax (row r+8*hl lives across the 16 lanes of each half)
    float fac8[8];
#pragma unroll
    for (int r = 0; r < 8; ++r) {
      float s0 = st0[r] * 0.125f;   // 1/sqrt(DK)
      float s1 = st1[r] * 0.125f;
      float mx = fmaxf(s0, s1);
#pragma unroll
      for (int off = 8; off > 0; off >>= 1)
        mx = fmaxf(mx, __shfl_xor(mx, off, 16));
      float mnew = fmaxf(ms[r], mx);
      float p0 = __expf(s0 - mnew);
      float p1 = __expf(s1 - mnew);
      float fac = __expf(ms[r] - mnew);
      float rsum = p0 + p1;
#pragma unroll
      for (int off = 8; off > 0; off >>= 1)
        rsum += __shfl_xor(rsum, off, 16);
      ls[r] = ls[r] * fac + rsum;
      ms[r] = mnew;
      fac8[r] = fac;
      // C-layout -> row-major P tile in per-wave LDS (A-fragment reload)
      plds[wave][r + 8 * hl][ln]      = (_Float16)p0;
      plds[wave][r + 8 * hl][16 + ln] = (_Float16)p1;
    }
#pragma unroll
    for (int t = 0; t < 4; ++t)
#pragma unroll
      for (int r = 0; r < 8; ++r)
        acc[t][r] *= fac8[r];
    // LDS ops are in-order within a wave (DScnt) — reload P as A fragment
    v16h pf = load_frag_a(&plds[wave][ln][hl * 16]);
    // --- P(16x32) x V(32x64): 4 WMMA tiles over dk
#pragma unroll
    for (int t = 0; t < 4; ++t)
      acc[t] = wmma_f16(pf, load_frag_b(&Vsm[cur][t * 16 + ln][hl * 16]),
                        acc[t]);
    __syncthreads();   // everyone done with this K/V stage before reload
  }

  float inv8[8];
#pragma unroll
  for (int r = 0; r < 8; ++r) inv8[r] = 1.f / ls[r];
  _Float16* crow = ctx + ((size_t)b * S_ + q0) * D_ + h * DK_;
#pragma unroll
  for (int t = 0; t < 4; ++t)
#pragma unroll
    for (int r = 0; r < 8; ++r)
      crow[(size_t)(r + 8 * hl) * D_ + t * 16 + ln] =
          (_Float16)(acc[t][r] * inv8[r]);
}

// ---------------------------------------------------------------------------
// Host-side orchestration
// ---------------------------------------------------------------------------
extern "C" void kernel_launch(void* const* d_in, const int* in_sizes, int n_in,
                              void* d_out, int out_size, void* d_ws,
                              size_t ws_size, hipStream_t stream) {
  const float* x   = (const float*)d_in[0];
  const float* Wq  = (const float*)d_in[1];
  const float* bq  = (const float*)d_in[2];
  const float* Wk  = (const float*)d_in[3];
  const float* bk  = (const float*)d_in[4];
  const float* Wv  = (const float*)d_in[5];
  const float* bv  = (const float*)d_in[6];
  const float* Wo  = (const float*)d_in[7];
  const float* bo  = (const float*)d_in[8];
  const float* W1  = (const float*)d_in[9];
  const float* b1  = (const float*)d_in[10];
  const float* W2  = (const float*)d_in[11];
  const float* b2  = (const float*)d_in[12];
  const float* g1  = (const float*)d_in[13];
  const float* be1 = (const float*)d_in[14];
  const float* g2  = (const float*)d_in[15];
  const float* be2 = (const float*)d_in[16];

  // Workspace carve-up (256B aligned)
  char* wp = (char*)d_ws;
  auto take = [&](size_t bytes) -> void* {
    void* p = (void*)wp;
    wp += (bytes + 255) & ~(size_t)255;
    return p;
  };
  _Float16* hbf  = (_Float16*)take((size_t)MTOK * D_ * 2);     // LN1(x)
  _Float16* Wqt  = (_Float16*)take((size_t)D_ * D_ * 2);
  _Float16* Wkt  = (_Float16*)take((size_t)D_ * D_ * 2);
  _Float16* Wvt  = (_Float16*)take((size_t)D_ * D_ * 2);
  _Float16* Wot  = (_Float16*)take((size_t)D_ * D_ * 2);
  _Float16* W1t  = (_Float16*)take((size_t)DFF_ * D_ * 2);
  _Float16* W2t  = (_Float16*)take((size_t)D_ * DFF_ * 2);
  _Float16* qbf  = (_Float16*)take((size_t)MTOK * D_ * 2);     // [B,H,S,DK]
  _Float16* kbf  = (_Float16*)take((size_t)MTOK * D_ * 2);     // [B,H,S,DK]
  _Float16* vTt  = (_Float16*)take((size_t)MTOK * D_ * 2);     // [B,H,DK,S]
  _Float16* ctxb = (_Float16*)take((size_t)MTOK * D_ * 2);     // [B,S,D]
  float*    x1   = (float*)   take((size_t)MTOK * D_ * 4);     // residual 1
  _Float16* h2bf = (_Float16*)take((size_t)MTOK * D_ * 2);     // LN2(x1)
  _Float16* ffbf = (_Float16*)take((size_t)MTOK * DFF_ * 2);   // relu(h2 W1+b1)

  // 1) Weight conversions (fp32 -> f16, transposed to [N,K])
  {
    int blks = (int)(((size_t)D_ * D_ + 255) / 256);
    qkv_w_transpose_f16<<<blks, 256, 0, stream>>>(Wq, Wqt);
    qkv_w_transpose_f16<<<blks, 256, 0, stream>>>(Wk, Wkt);
    qkv_w_transpose_f16<<<blks, 256, 0, stream>>>(Wv, Wvt);
    w_transpose_f16<<<blks, 256, 0, stream>>>(Wo, Wot, D_, D_);
    int blksF = (int)(((size_t)DFF_ * D_ + 255) / 256);
    w_transpose_f16<<<blksF, 256, 0, stream>>>(W1, W1t, D_, DFF_);
    w_transpose_f16<<<blksF, 256, 0, stream>>>(W2, W2t, DFF_, D_);
  }

  // 2) LN1
  ln_f16_kernel<<<MTOK, 256, 0, stream>>>(x, g1, be1, hbf);

  // 3) QKV projections (M=4096, N=1024, K=1024)
  {
    dim3 grid(D_ / 64, MTOK / 128);
    gemm_f16_kernel<EPI_QK><<<grid, 256, 0, stream>>>(hbf, Wqt, bq, nullptr,
                                                      qbf, MTOK, D_, D_);
    gemm_f16_kernel<EPI_QK><<<grid, 256, 0, stream>>>(hbf, Wkt, bk, nullptr,
                                                      kbf, MTOK, D_, D_);
    gemm_f16_kernel<EPI_VT><<<grid, 256, 0, stream>>>(hbf, Wvt, bv, nullptr,
                                                      vTt, MTOK, D_, D_);
  }

  // 4) Flash attention -> ctx [B,S,D]
  {
    dim3 grid(S_ / 64, B_ * H_);
    flash_attn_kernel<<<grid, 128, 0, stream>>>(qbf, kbf, vTt, ctxb);
  }

  // 5) Output projection + residual: x1 = x + ctx*Wo + bo
  {
    dim3 grid(D_ / 64, MTOK / 128);
    gemm_f16_kernel<EPI_RES><<<grid, 256, 0, stream>>>(ctxb, Wot, bo, x, x1,
                                                       MTOK, D_, D_);
  }

  // 6) LN2
  ln_f16_kernel<<<MTOK, 256, 0, stream>>>(x1, g2, be2, h2bf);

  // 7) FFN up: ff = relu(h2*W1 + b1)  (N=4096, K=1024)
  {
    dim3 grid(DFF_ / 64, MTOK / 128);
    gemm_f16_kernel<EPI_RELU><<<grid, 256, 0, stream>>>(h2bf, W1t, b1, nullptr,
                                                        ffbf, MTOK, DFF_, D_);
  }

  // 8) FFN down + residual: out = x1 + ff*W2 + b2  (N=1024, K=4096)
  {
    dim3 grid(D_ / 64, MTOK / 128);
    gemm_f16_kernel<EPI_RES><<<grid, 256, 0, stream>>>(ffbf, W2t, b2, x1,
                                                       (float*)d_out, MTOK, D_,
                                                       DFF_);
  }
}
